// DiTBlock_88897233093043
// MI455X (gfx1250) — matrix operand order repaired and checked by reading, not compiled, not run on hardware
//
#include <hip/hip_runtime.h>
#include <hip/hip_bf16.h>

// ---------------------------------------------------------------------------
// DiT block for MI455X (gfx1250): bf16 WMMA GEMMs with fp32 accumulation.
// B=4, N=2048, C=768, H=12, D=64, FFN=2048.
// ---------------------------------------------------------------------------

typedef __bf16 bf16;
typedef __attribute__((ext_vector_type(16))) __bf16 v16bf;
typedef __attribute__((ext_vector_type(8)))  float  v8f;

#define BDIM 4
#define NDIM 2048
#define CDIM 768
#define HDIM 12
#define DDIM 64
#define FFNDIM 2048
#define MROWS (BDIM * NDIM)   // 8192
#define SIXC (6 * CDIM)       // 4608

union FragU { v16bf v; uint4 q[2]; };

__device__ __forceinline__ bf16 f2bf(float f) {
    unsigned u = __float_as_uint(f);
    u += 0x7fffu + ((u >> 16) & 1u);          // round-to-nearest-even
    unsigned short s = (unsigned short)(u >> 16);
    return __builtin_bit_cast(bf16, s);
}

// A-matrix fragment, 16x32 bf16 (M x K), row-major source with leading dim ld.
// lane 0-15: row M=lane, K = {k0..k0+7, k0+16..k0+23}
// lane 16-31: row M=lane-16, K = {k0+8..k0+15, k0+24..k0+31}
__device__ __forceinline__ v16bf load_a_frag(const bf16* __restrict__ base, int ld,
                                             int row0, int k0, int lane) {
    int r  = row0 + (lane & 15);
    int kb = k0 + ((lane & 16) ? 8 : 0);
    const bf16* p = base + (size_t)r * ld + kb;
    FragU f;
    f.q[0] = *reinterpret_cast<const uint4*>(p);
    f.q[1] = *reinterpret_cast<const uint4*>(p + 16);
    return f.v;
}

// B-matrix fragment, 32x16 bf16 (K x N). Source stored "column rows" contiguous
// in K (i.e. W[n, k] row-major => column n of B).
// lane 0-15: col N=lane, K = {k0..k0+7, k0+16..k0+23} ; lane 16-31: col N=lane-16, K = {k0+8..k0+15, k0+24..k0+31}
__device__ __forceinline__ v16bf load_b_frag(const bf16* __restrict__ base, int ld,
                                             int col0, int k0, int lane) {
    int r  = col0 + (lane & 15);
    int kb = k0 + ((lane & 16) ? 8 : 0);
    const bf16* p = base + (size_t)r * ld + kb;
    FragU f;
    f.q[0] = *reinterpret_cast<const uint4*>(p);
    f.q[1] = *reinterpret_cast<const uint4*>(p + 16);
    return f.v;
}

__device__ __forceinline__ v8f wmma_bf16(v16bf a, v16bf b, v8f c) {
    return __builtin_amdgcn_wmma_f32_16x16x32_bf16(false, a, false, b,
                                                   (short)0, c, false, false);
}

__device__ __forceinline__ v8f vzero8() {
    v8f z = {0.f, 0.f, 0.f, 0.f, 0.f, 0.f, 0.f, 0.f};
    return z;
}

// ---------------------------------------------------------------------------
// K0: mod = silu(c) @ adaln_w^T + adaln_b    [B, 6C]
// one wave per output element, lane-strided dot over C=768.
// ---------------------------------------------------------------------------
__global__ __launch_bounds__(256) void adaln_kernel(
    const float* __restrict__ c, const float* __restrict__ w,
    const float* __restrict__ bvec, float* __restrict__ mod) {
    int gw = (blockIdx.x * 256 + threadIdx.x) >> 5;   // wave id, 0 .. B*6C-1
    int ln = threadIdx.x & 31;
    int j = gw % SIXC, b = gw / SIXC;
    const float* cr = c + (size_t)b * CDIM;
    const float* wr = w + (size_t)j * CDIM;
    float acc = 0.f;
    for (int k = ln; k < CDIM; k += 32) {
        float cv = cr[k];
        acc += (cv / (1.f + __expf(-cv))) * wr[k];
    }
    for (int m = 16; m >= 1; m >>= 1) acc += __shfl_xor(acc, m, 32);
    if (ln == 0) mod[(size_t)b * SIXC + j] = acc + bvec[j];
}

// ---------------------------------------------------------------------------
// fp32 -> bf16 elementwise conversion (weights)
// ---------------------------------------------------------------------------
__global__ __launch_bounds__(256) void cvt_bf16(const float* __restrict__ s,
                                                bf16* __restrict__ d, int n) {
    int i = blockIdx.x * 256 + threadIdx.x;
    if (i < n) d[i] = f2bf(s[i]);
}

// ---------------------------------------------------------------------------
// K2/K7: out_bf16 = modulate(LayerNorm(x), shift, scale). One block per row.
// ---------------------------------------------------------------------------
__global__ __launch_bounds__(256) void ln_mod_kernel(
    const float* __restrict__ x, const float* __restrict__ mod,
    int shift_off, int scale_off, bf16* __restrict__ out) {
    int row = blockIdx.x;                 // 0..MROWS-1
    int b = row >> 11;                    // row / N
    const float* xr = x + (size_t)row * CDIM;
    int t = threadIdx.x;
    float v0 = xr[t], v1 = xr[t + 256], v2 = xr[t + 512];
    float s = v0 + v1 + v2;
    float ss = v0 * v0 + v1 * v1 + v2 * v2;
    for (int m = 16; m >= 1; m >>= 1) {
        s  += __shfl_xor(s, m, 32);
        ss += __shfl_xor(ss, m, 32);
    }
    __shared__ float sbuf[8], ssbuf[8];
    int wv = t >> 5, ln = t & 31;
    if (ln == 0) { sbuf[wv] = s; ssbuf[wv] = ss; }
    __syncthreads();
    float tot = 0.f, tots = 0.f;
    for (int i = 0; i < 8; i++) { tot += sbuf[i]; tots += ssbuf[i]; }
    float mean = tot * (1.f / CDIM);
    float var  = tots * (1.f / CDIM) - mean * mean;
    float inv  = rsqrtf(var + 1e-6f);
    const float* sh = mod + (size_t)b * SIXC + shift_off;
    const float* sc = mod + (size_t)b * SIXC + scale_off;
    bf16* orow = out + (size_t)row * CDIM;
    orow[t]       = f2bf((v0 - mean) * inv * (1.f + sc[t])       + sh[t]);
    orow[t + 256] = f2bf((v1 - mean) * inv * (1.f + sc[t + 256]) + sh[t + 256]);
    orow[t + 512] = f2bf((v2 - mean) * inv * (1.f + sc[t + 512]) + sh[t + 512]);
}

// ---------------------------------------------------------------------------
// WMMA GEMM: out[M,Nc] = A[M,K] (bf16) x W[Nc,K]^T (bf16), fp32 accumulate.
// fuse==0: out = acc (+bias[n] if bias)
// fuse==1: out = residual + mod[b,gate_off+n] * (acc + bias[n])
// Each wave: 32x64 tile (8 wmma per K=32). Block: 8 waves => 64x256 tile.
// ---------------------------------------------------------------------------
__global__ __launch_bounds__(256) void gemm_kernel(
    const bf16* __restrict__ A, const bf16* __restrict__ W,
    const float* __restrict__ bias, const float* __restrict__ residual,
    const float* __restrict__ mod, int gate_off, int rows_per_b,
    float* __restrict__ out, int M, int Nc, int K, int fuse) {
    int lane = threadIdx.x & 31;
    int wave = threadIdx.x >> 5;
    int m0 = blockIdx.y * 64 + (wave & 1) * 32;
    int n0 = blockIdx.x * 256 + (wave >> 1) * 64;

    v8f acc[2][4];
    for (int i = 0; i < 2; i++)
        for (int j = 0; j < 4; j++) acc[i][j] = vzero8();

    for (int k = 0; k < K; k += 32) {
        v16bf a0 = load_a_frag(A, K, m0,      k, lane);
        v16bf a1 = load_a_frag(A, K, m0 + 16, k, lane);
        v16bf b0 = load_b_frag(W, K, n0,      k, lane);
        v16bf b1 = load_b_frag(W, K, n0 + 16, k, lane);
        v16bf b2 = load_b_frag(W, K, n0 + 32, k, lane);
        v16bf b3 = load_b_frag(W, K, n0 + 48, k, lane);
        acc[0][0] = wmma_bf16(a0, b0, acc[0][0]);
        acc[0][1] = wmma_bf16(a0, b1, acc[0][1]);
        acc[0][2] = wmma_bf16(a0, b2, acc[0][2]);
        acc[0][3] = wmma_bf16(a0, b3, acc[0][3]);
        acc[1][0] = wmma_bf16(a1, b0, acc[1][0]);
        acc[1][1] = wmma_bf16(a1, b1, acc[1][1]);
        acc[1][2] = wmma_bf16(a1, b2, acc[1][2]);
        acc[1][3] = wmma_bf16(a1, b3, acc[1][3]);
    }

    int hl = lane >> 4, col = lane & 15;
    for (int i = 0; i < 2; i++) {
        for (int j = 0; j < 4; j++) {
            for (int r = 0; r < 8; r++) {
                int m = m0 + 16 * i + r + 8 * hl;
                int n = n0 + 16 * j + col;
                float v = acc[i][j][r];
                if (bias) v += bias[n];
                size_t oi = (size_t)m * Nc + n;
                if (fuse) {
                    int bb = m / rows_per_b;
                    v = residual[oi] + mod[(size_t)bb * SIXC + gate_off + n] * v;
                }
                out[oi] = v;
            }
        }
    }
}

// ---------------------------------------------------------------------------
// K4: rmsnorm q/k, pack q,k [B,H,N,D] bf16 and v transposed [B,H,D,N] bf16.
// One wave per (row m, head h); 32 lanes x 2 elements over D=64.
// ---------------------------------------------------------------------------
__global__ __launch_bounds__(256) void qkv_pack_kernel(
    const float* __restrict__ qkv, const float* __restrict__ qs,
    const float* __restrict__ ks, bf16* __restrict__ qb,
    bf16* __restrict__ kb, bf16* __restrict__ vT) {
    int gw = (blockIdx.x * 256 + threadIdx.x) >> 5;   // 0 .. MROWS*H-1
    int ln = threadIdx.x & 31;
    int h = gw % HDIM, m = gw / HDIM;
    int b = m >> 11, n = m & (NDIM - 1);
    const float* base = qkv + (size_t)m * (3 * CDIM) + h * DDIM;
    int d0 = ln * 2;

    // q
    float a0 = base[d0], a1 = base[d0 + 1];
    float sum = a0 * a0 + a1 * a1;
    for (int msk = 16; msk >= 1; msk >>= 1) sum += __shfl_xor(sum, msk, 32);
    float inv = 1.f / (sqrtf(sum * (1.f / DDIM)) + 1e-8f);
    size_t qidx = (((size_t)(b * HDIM + h) * NDIM) + n) * DDIM + d0;
    qb[qidx]     = f2bf(qs[d0] * a0 * inv);
    qb[qidx + 1] = f2bf(qs[d0 + 1] * a1 * inv);

    // k
    a0 = base[CDIM + d0]; a1 = base[CDIM + d0 + 1];
    sum = a0 * a0 + a1 * a1;
    for (int msk = 16; msk >= 1; msk >>= 1) sum += __shfl_xor(sum, msk, 32);
    inv = 1.f / (sqrtf(sum * (1.f / DDIM)) + 1e-8f);
    kb[qidx]     = f2bf(ks[d0] * a0 * inv);
    kb[qidx + 1] = f2bf(ks[d0 + 1] * a1 * inv);

    // v -> transposed [B,H,D,N]
    a0 = base[2 * CDIM + d0]; a1 = base[2 * CDIM + d0 + 1];
    size_t vbase = (size_t)(b * HDIM + h) * DDIM;
    vT[(vbase + d0)     * NDIM + n] = f2bf(a0);
    vT[(vbase + d0 + 1) * NDIM + n] = f2bf(a1);
}

// ---------------------------------------------------------------------------
// K5: flash attention. Block = (b,h, 128 queries), 8 waves x 16 queries.
// Per 32-key chunk: 4 wmma for S = Q K^T, online softmax, P staged via LDS,
// 4 wmma for O += P V (V pre-transposed).  Output written as [B,N,H*D] bf16.
// ---------------------------------------------------------------------------
__global__ __launch_bounds__(256) void attn_kernel(
    const bf16* __restrict__ qb, const bf16* __restrict__ kb,
    const bf16* __restrict__ vT, const float* __restrict__ bias,
    bf16* __restrict__ ob) {
    __shared__ bf16 P[8][16 * 32];
    int lane = threadIdx.x & 31;
    int wave = threadIdx.x >> 5;
    int bh = blockIdx.x;                 // 0..B*H-1
    int b = bh / HDIM, h = bh % HDIM;
    int q0 = blockIdx.y * 128 + wave * 16;
    const bf16* Q  = qb + (size_t)bh * NDIM * DDIM;
    const bf16* Kp = kb + (size_t)bh * NDIM * DDIM;
    const bf16* V  = vT + (size_t)bh * DDIM * NDIM;
    const float* Bs = bias + (size_t)h * NDIM * NDIM;
    int hl = lane >> 4, col = lane & 15;
    bf16* pw = &P[wave][0];

    v16bf qa0 = load_a_frag(Q, DDIM, q0, 0,  lane);
    v16bf qa1 = load_a_frag(Q, DDIM, q0, 32, lane);

    v8f o0 = vzero8(), o1 = vzero8(), o2 = vzero8(), o3 = vzero8();
    float mr[8], lr[8];
    for (int r = 0; r < 8; r++) { mr[r] = -3.0e38f; lr[r] = 0.f; }

    for (int j = 0; j < NDIM; j += 32) {
        v8f s0 = vzero8(), s1 = vzero8();
        {
            v16bf k0f = load_b_frag(Kp, DDIM, j, 0,  lane);
            v16bf k1f = load_b_frag(Kp, DDIM, j, 32, lane);
            s0 = wmma_bf16(qa0, k0f, s0);
            s0 = wmma_bf16(qa1, k1f, s0);
            k0f = load_b_frag(Kp, DDIM, j + 16, 0,  lane);
            k1f = load_b_frag(Kp, DDIM, j + 16, 32, lane);
            s1 = wmma_bf16(qa0, k0f, s1);
            s1 = wmma_bf16(qa1, k1f, s1);
        }
        for (int r = 0; r < 8; r++) {
            int qm = q0 + r + 8 * hl;
            const float* bp = Bs + (size_t)qm * NDIM + j + col;
            s0[r] = s0[r] * 0.125f + bp[0];
            s1[r] = s1[r] * 0.125f + bp[16];
            float mx = fmaxf(s0[r], s1[r]);
            for (int msk = 8; msk >= 1; msk >>= 1)
                mx = fmaxf(mx, __shfl_xor(mx, msk, 32));   // within 16-lane half
            mx = fmaxf(mx, mr[r]);
            float scale = __expf(mr[r] - mx);
            mr[r] = mx;
            float p0 = __expf(s0[r] - mx);
            float p1 = __expf(s1[r] - mx);
            float rs = p0 + p1;
            for (int msk = 8; msk >= 1; msk >>= 1)
                rs += __shfl_xor(rs, msk, 32);
            lr[r] = lr[r] * scale + rs;
            o0[r] *= scale; o1[r] *= scale; o2[r] *= scale; o3[r] *= scale;
            int row = r + 8 * hl;
            pw[row * 32 + col]      = f2bf(p0);
            pw[row * 32 + 16 + col] = f2bf(p1);
        }
        // per-wave private LDS slot; DS ops are in-order, make the dependency
        // explicit with a CDNA5 split-counter wait before re-reading.
        asm volatile("s_wait_dscnt 0" ::: "memory");
        v16bf pa = load_a_frag(pw, 32, 0, 0, lane);
        v16bf vb0 = load_b_frag(V, NDIM, 0,  j, lane);
        v16bf vb1 = load_b_frag(V, NDIM, 16, j, lane);
        v16bf vb2 = load_b_frag(V, NDIM, 32, j, lane);
        v16bf vb3 = load_b_frag(V, NDIM, 48, j, lane);
        o0 = wmma_bf16(pa, vb0, o0);
        o1 = wmma_bf16(pa, vb1, o1);
        o2 = wmma_bf16(pa, vb2, o2);
        o3 = wmma_bf16(pa, vb3, o3);
    }

    for (int r = 0; r < 8; r++) {
        float inv = 1.f / lr[r];
        int qm = q0 + r + 8 * hl;
        size_t base = ((size_t)(b * NDIM + qm)) * CDIM + h * DDIM + col;
        ob[base]      = f2bf(o0[r] * inv);
        ob[base + 16] = f2bf(o1[r] * inv);
        ob[base + 32] = f2bf(o2[r] * inv);
        ob[base + 48] = f2bf(o3[r] * inv);
    }
}

// ---------------------------------------------------------------------------
// K10: g = bf16( silu(u1) * u3 )
// ---------------------------------------------------------------------------
__global__ __launch_bounds__(256) void silu_mul_kernel(
    const float* __restrict__ u1, const float* __restrict__ u3,
    bf16* __restrict__ g, int n) {
    int i = blockIdx.x * 256 + threadIdx.x;
    if (i < n) {
        float a = u1[i];
        g[i] = f2bf((a / (1.f + __expf(-a))) * u3[i]);
    }
}

// ---------------------------------------------------------------------------
extern "C" void kernel_launch(void* const* d_in, const int* in_sizes, int n_in,
                              void* d_out, int out_size, void* d_ws, size_t ws_size,
                              hipStream_t stream) {
    const float* x       = (const float*)d_in[0];
    const float* c       = (const float*)d_in[1];
    const float* bias    = (const float*)d_in[2];
    const float* qkv_w   = (const float*)d_in[3];
    const float* qkv_b   = (const float*)d_in[4];
    const float* q_scale = (const float*)d_in[5];
    const float* k_scale = (const float*)d_in[6];
    const float* proj_w  = (const float*)d_in[7];
    const float* proj_b  = (const float*)d_in[8];
    const float* w1      = (const float*)d_in[9];
    const float* w2_w    = (const float*)d_in[10];
    const float* w2_b    = (const float*)d_in[11];
    const float* w3      = (const float*)d_in[12];
    const float* adaln_w = (const float*)d_in[13];
    const float* adaln_b = (const float*)d_in[14];
    float* out = (float*)d_out;

    // workspace layout (256B aligned slabs)
    char* wp = (char*)d_ws;
    auto alloc = [&](size_t bytes) {
        char* p = wp;
        wp += (bytes + 255) & ~(size_t)255;
        return p;
    };
    float* mod    = (float*)alloc((size_t)BDIM * SIXC * 4);
    bf16*  wqkv   = (bf16*)alloc((size_t)3 * CDIM * CDIM * 2);
    bf16*  wproj  = (bf16*)alloc((size_t)CDIM * CDIM * 2);
    bf16*  w1b    = (bf16*)alloc((size_t)FFNDIM * CDIM * 2);
    bf16*  w3b    = (bf16*)alloc((size_t)FFNDIM * CDIM * 2);
    bf16*  w2b    = (bf16*)alloc((size_t)CDIM * FFNDIM * 2);
    bf16*  hbuf   = (bf16*)alloc((size_t)MROWS * CDIM * 2);        // h1 -> o -> h2
    float* big0   = (float*)alloc((size_t)MROWS * 3 * CDIM * 4);   // qkv -> u1
    bf16*  qb     = (bf16*)alloc((size_t)MROWS * CDIM * 2);
    bf16*  kb     = (bf16*)alloc((size_t)MROWS * CDIM * 2);
    bf16*  vT     = (bf16*)alloc((size_t)MROWS * CDIM * 2);
    float* x1     = (float*)alloc((size_t)MROWS * CDIM * 4);
    float* u3     = (float*)alloc((size_t)MROWS * FFNDIM * 4);
    bf16*  gbuf   = (bf16*)alloc((size_t)MROWS * FFNDIM * 2);
    (void)ws_size; (void)n_in; (void)in_sizes; (void)out_size;

    // K0: adaLN modulation vector
    adaln_kernel<<<dim3((BDIM * SIXC * 32) / 256), dim3(256), 0, stream>>>(
        c, adaln_w, adaln_b, mod);

    // weights -> bf16
    cvt_bf16<<<dim3((3 * CDIM * CDIM + 255) / 256), dim3(256), 0, stream>>>(qkv_w, wqkv, 3 * CDIM * CDIM);
    cvt_bf16<<<dim3((CDIM * CDIM + 255) / 256),     dim3(256), 0, stream>>>(proj_w, wproj, CDIM * CDIM);
    cvt_bf16<<<dim3((FFNDIM * CDIM + 255) / 256),   dim3(256), 0, stream>>>(w1, w1b, FFNDIM * CDIM);
    cvt_bf16<<<dim3((FFNDIM * CDIM + 255) / 256),   dim3(256), 0, stream>>>(w3, w3b, FFNDIM * CDIM);
    cvt_bf16<<<dim3((CDIM * FFNDIM + 255) / 256),   dim3(256), 0, stream>>>(w2_w, w2b, CDIM * FFNDIM);

    // K2: h1 = modulate(LN(x), shift_msa, scale_msa)
    ln_mod_kernel<<<dim3(MROWS), dim3(256), 0, stream>>>(x, mod, 0, CDIM, hbuf);

    // K3: qkv = h1 @ qkv_w^T + qkv_b   [M, 3C]
    gemm_kernel<<<dim3(3 * CDIM / 256, MROWS / 64), dim3(256), 0, stream>>>(
        hbuf, wqkv, qkv_b, nullptr, nullptr, 0, NDIM, big0, MROWS, 3 * CDIM, CDIM, 0);

    // K4: rmsnorm + pack q,k,[vT]
    qkv_pack_kernel<<<dim3(MROWS * HDIM * 32 / 256), dim3(256), 0, stream>>>(
        big0, q_scale, k_scale, qb, kb, vT);

    // K5: attention -> o (bf16, [B,N,H*D]) into hbuf
    attn_kernel<<<dim3(BDIM * HDIM, NDIM / 128), dim3(256), 0, stream>>>(
        qb, kb, vT, bias, hbuf);

    // K6: x1 = x + gate_msa * (o @ proj_w^T + proj_b)
    gemm_kernel<<<dim3(CDIM / 256, MROWS / 64), dim3(256), 0, stream>>>(
        hbuf, wproj, proj_b, x, mod, 2 * CDIM, NDIM, x1, MROWS, CDIM, CDIM, 1);

    // K7: h2 = modulate(LN(x1), shift_mlp, scale_mlp)
    ln_mod_kernel<<<dim3(MROWS), dim3(256), 0, stream>>>(x1, mod, 3 * CDIM, 4 * CDIM, hbuf);

    // K8/K9: u1 = h2 @ w1^T ; u3 = h2 @ w3^T
    gemm_kernel<<<dim3(FFNDIM / 256, MROWS / 64), dim3(256), 0, stream>>>(
        hbuf, w1b, nullptr, nullptr, nullptr, 0, NDIM, big0, MROWS, FFNDIM, CDIM, 0);
    gemm_kernel<<<dim3(FFNDIM / 256, MROWS / 64), dim3(256), 0, stream>>>(
        hbuf, w3b, nullptr, nullptr, nullptr, 0, NDIM, u3, MROWS, FFNDIM, CDIM, 0);

    // K10: g = silu(u1) * u3
    silu_mul_kernel<<<dim3(MROWS * FFNDIM / 256), dim3(256), 0, stream>>>(
        big0, u3, gbuf, MROWS * FFNDIM);

    // K11: out = x1 + gate_mlp * (g @ w2^T + w2_b)
    gemm_kernel<<<dim3(CDIM / 256, MROWS / 64), dim3(256), 0, stream>>>(
        gbuf, w2b, w2_b, x1, mod, 5 * CDIM, NDIM, out, MROWS, CDIM, FFNDIM, 1);
}
